// Qwen3MoeSparseMoeBlock_36283883717171
// MI455X (gfx1250) — matrix operand order repaired and checked by reading, not compiled, not run on hardware
//
#include <hip/hip_runtime.h>
#include <hip/hip_bf16.h>

// Problem constants (match reference): H=1024, I=768, E=16, K=2, B=2, S=2048
#define HDIM 1024
#define IDIM 768
#define NEXP 16
#define TOPK 2
#define TOKENS 4096
#define MROWS 32   // token rows per workgroup = 2 WMMA M-tiles

typedef __attribute__((ext_vector_type(16))) __bf16 v16bf;
typedef __attribute__((ext_vector_type(8)))  float  v8f;

union Frag { unsigned int u[8]; uint4 q[2]; v16bf v; };

__device__ __forceinline__ unsigned short f2bf(float f) {
  unsigned int u = __float_as_uint(f);
  u += 0x7FFFu + ((u >> 16) & 1u);   // round-to-nearest-even
  return (unsigned short)(u >> 16);
}
__device__ __forceinline__ unsigned int packbf2(float lo, float hi) {
  return (unsigned int)f2bf(lo) | ((unsigned int)f2bf(hi) << 16);
}

// ---- WMMA fragment loads (bf16 already in memory; pure b128 loads) ----
// A (16x32 MxK): lane = half*16 + m; VGPR i holds K = (i/4)*16 + half*8 + (i%4)*2 + {0,1}
//   -> two contiguous 16B runs: [k0 + half*8, +8) and [k0 + 16 + half*8, +8)
__device__ __forceinline__ v16bf load_a_glb(const unsigned short* __restrict__ row,
                                            int k0, int half) {
  const unsigned short* p = row + k0 + (half << 3);
  Frag r;
  r.q[0] = *(const uint4*)p;
  r.q[1] = *(const uint4*)(p + 16);
  return r.v;
}
__device__ __forceinline__ v16bf load_a_lds(const unsigned short* lds, int ldk,
                                            int k0, int lane) {
  const int m = lane & 15, half = lane >> 4;
  const unsigned short* p = lds + m * ldk + k0 + (half << 3);
  Frag r;
  r.q[0] = *(const uint4*)p;
  r.q[1] = *(const uint4*)(p + 16);
  return r.v;
}
// B (32x16 KxN): lane = half*16 + n; VGPR i holds K = (i/4)*16 + half*8 + (i%4)*2 + {0,1}
//   -> two contiguous 16B runs: [k0 + half*8, +8) and [k0 + 16 + half*8, +8)
__device__ __forceinline__ v16bf load_b_glb(const unsigned short* __restrict__ row,
                                            int k0, int half) {
  const unsigned short* p = row + k0 + (half << 3);
  Frag r;
  r.q[0] = *(const uint4*)p;
  r.q[1] = *(const uint4*)(p + 16);
  return r.v;
}

// ---------------- fp32 -> bf16 streaming convert (8 elems/thread) ----------------
__global__ void cvt_bf16_kernel(const float* __restrict__ src,
                                unsigned short* __restrict__ dst, int n8) {
  for (int i = blockIdx.x * blockDim.x + threadIdx.x; i < n8; i += gridDim.x * blockDim.x) {
    const float4 a = ((const float4*)src)[2 * i];
    const float4 b = ((const float4*)src)[2 * i + 1];
    uint4 o;
    o.x = packbf2(a.x, a.y); o.y = packbf2(a.z, a.w);
    o.z = packbf2(b.x, b.y); o.w = packbf2(b.z, b.w);
    ((uint4*)dst)[i] = o;
  }
}

// ---------------- init ----------------
__global__ void zero_out_kernel(float* __restrict__ out, int n) {
  for (int i = blockIdx.x * blockDim.x + threadIdx.x; i < n; i += gridDim.x * blockDim.x)
    out[i] = 0.f;
}
// zero router counts + the zero-pad activation row (row index TOKENS of xbf)
__global__ void init_misc_kernel(int* __restrict__ counts,
                                 unsigned short* __restrict__ xbf_pad_row) {
  const int tid = threadIdx.x;                 // 256 threads
  if (tid < NEXP) counts[tid] = 0;
  ((uint2*)xbf_pad_row)[tid] = make_uint2(0u, 0u);   // 256 * 8B = 2KB row
}

// ---------------- router: logits -> top-2 -> renormalized weights ----------------
__global__ __launch_bounds__(128)
void router_kernel(const float* __restrict__ x, const float* __restrict__ gw,
                   int* __restrict__ counts, int* __restrict__ topIdx,
                   float* __restrict__ topW) {
  __shared__ float red[128 * NEXP];
  const int t = blockIdx.x;
  const int tid = threadIdx.x;
  float acc[NEXP];
#pragma unroll
  for (int e = 0; e < NEXP; ++e) acc[e] = 0.f;
  const float* xr = x + (size_t)t * HDIM;
  for (int h = tid; h < HDIM; h += 128) {
    const float xv = xr[h];
#pragma unroll
    for (int e = 0; e < NEXP; ++e) acc[e] += xv * gw[e * HDIM + h];
  }
#pragma unroll
  for (int e = 0; e < NEXP; ++e) red[tid * NEXP + e] = acc[e];
  __syncthreads();
  if (tid < NEXP) {
    float s = 0.f;
    for (int i = 0; i < 128; ++i) s += red[i * NEXP + tid];
    red[tid] = s;
  }
  __syncthreads();
  if (tid == 0) {
    int e0 = 0; float l0 = red[0];
    for (int e = 1; e < NEXP; ++e) if (red[e] > l0) { l0 = red[e]; e0 = e; }
    int e1 = (e0 == 0) ? 1 : 0; float l1 = red[e1];
    for (int e = 0; e < NEXP; ++e) if (e != e0 && red[e] > l1) { l1 = red[e]; e1 = e; }
    const float p1 = __expf(l1 - l0);           // softmax denom cancels in renorm
    const float inv = 1.f / (1.f + p1);
    topIdx[t * 2 + 0] = e0; topW[t * 2 + 0] = inv;
    topIdx[t * 2 + 1] = e1; topW[t * 2 + 1] = p1 * inv;
    atomicAdd(&counts[e0], 1);
    atomicAdd(&counts[e1], 1);
  }
}

// ---------------- bucket tokens per expert ----------------
__global__ void scan_kernel(const int* __restrict__ counts, int* __restrict__ offsets,
                            int* __restrict__ cursor) {
  if (threadIdx.x == 0) {
    int s = 0;
    for (int e = 0; e < NEXP; ++e) { offsets[e] = s; cursor[e] = s; s += counts[e]; }
    offsets[NEXP] = s;
  }
}
__global__ void scatter_kernel(const int* __restrict__ topIdx, const float* __restrict__ topW,
                               int* __restrict__ cursor, int* __restrict__ tokIds,
                               float* __restrict__ tokW) {
  const int t = blockIdx.x * blockDim.x + threadIdx.x;
  if (t >= TOKENS) return;
#pragma unroll
  for (int k = 0; k < TOPK; ++k) {
    const int e = topIdx[t * 2 + k];
    const int pos = atomicAdd(&cursor[e], 1);
    tokIds[pos] = t;
    tokW[pos] = topW[t * 2 + k];
  }
}

// ---------------- expert SwiGLU, all-bf16 operands, branch-free WMMA inner loops ----------------
__global__ __launch_bounds__(256)
void moe_expert_kernel(const unsigned short* __restrict__ xbf,   // [T+1][H] bf16 (last row = zeros)
                       const unsigned short* __restrict__ Wg,    // [E][I][H] bf16
                       const unsigned short* __restrict__ Wu,    // [E][I][H] bf16
                       const unsigned short* __restrict__ Wd,    // [E][H][I] bf16
                       const int* __restrict__ counts,
                       const int* __restrict__ offsets,
                       const int* __restrict__ tokIds,
                       const float* __restrict__ tokW,
                       float* __restrict__ out) {
  __shared__ unsigned short sH[MROWS * IDIM];  // 48 KB: silu(g)*u, bf16
  __shared__ float sW[MROWS];
  __shared__ int   sTok[MROWS];

  const int e  = blockIdx.x;
  const int bm = blockIdx.y;
  const int cnt = counts[e];
  if (bm * MROWS >= cnt) return;
  const int off  = offsets[e];
  const int tid  = threadIdx.x;
  const int lane = tid & 31;
  const int wave = tid >> 5;
  const int half = lane >> 4;
  const int ml   = lane & 15;          // A-row / B-col index for this lane

  if (tid < MROWS) {
    const int r = bm * MROWS + tid;
    if (r < cnt) { sTok[tid] = tokIds[off + r]; sW[tid] = tokW[off + r]; }
    else         { sTok[tid] = -1;              sW[tid] = 0.f; }
  }
  __syncthreads();

  // invalid rows point at the dedicated all-zero pad row -> branch-free A loads
  const int t0 = sTok[ml], t1 = sTok[16 + ml];
  const unsigned short* aRow0 = xbf + (size_t)(t0 >= 0 ? t0 : TOKENS) * HDIM;
  const unsigned short* aRow1 = xbf + (size_t)(t1 >= 0 ? t1 : TOKENS) * HDIM;

  const unsigned short* Wge = Wg + (size_t)e * IDIM * HDIM;
  const unsigned short* Wue = Wu + (size_t)e * IDIM * HDIM;
  const unsigned short* Wde = Wd + (size_t)e * HDIM * IDIM;

  // Phase 1: g/u (48 N-tiles over I, 6 per wave); each B frag feeds 2 M-tiles.
  // K-loop kept rolled (x2) to bound VGPR pressure -> no scratch spills.
  for (int nt = wave; nt < IDIM / 16; nt += 8) {
    const unsigned short* bgRow = Wge + (size_t)(nt * 16 + ml) * HDIM;
    const unsigned short* buRow = Wue + (size_t)(nt * 16 + ml) * HDIM;
    v8f accG0 = {}, accU0 = {}, accG1 = {}, accU1 = {};
#pragma clang loop unroll_count(2)
    for (int k0 = 0; k0 < HDIM; k0 += 32) {
      const v16bf bg = load_b_glb(bgRow, k0, half);
      const v16bf bu = load_b_glb(buRow, k0, half);
      const v16bf a0 = load_a_glb(aRow0, k0, half);
      const v16bf a1 = load_a_glb(aRow1, k0, half);
      if (k0 + 64 < HDIM) {
        __builtin_prefetch(bgRow + k0 + 64, 0, 1);   // global_prefetch_b8
        __builtin_prefetch(buRow + k0 + 64, 0, 1);
      }
      accG0 = __builtin_amdgcn_wmma_f32_16x16x32_bf16(false, a0, false, bg, (short)0, accG0, false, false);
      accU0 = __builtin_amdgcn_wmma_f32_16x16x32_bf16(false, a0, false, bu, (short)0, accU0, false, false);
      accG1 = __builtin_amdgcn_wmma_f32_16x16x32_bf16(false, a1, false, bg, (short)0, accG1, false, false);
      accU1 = __builtin_amdgcn_wmma_f32_16x16x32_bf16(false, a1, false, bu, (short)0, accU1, false, false);
    }
    const int n = nt * 16 + ml;
#pragma unroll
    for (int j = 0; j < 8; ++j) {              // C/D layout: m = j + 8*half, n = lane%16
      const int m = j + 8 * half;
      float g = accG0[j], u = accU0[j];
      sH[m * IDIM + n] = f2bf(g / (1.f + __expf(-g)) * u);
      g = accG1[j]; u = accU1[j];
      sH[(16 + m) * IDIM + n] = f2bf(g / (1.f + __expf(-g)) * u);
    }
  }
  __syncthreads();

  // Phase 2: down-proj (64 N-tiles over H, 8 per wave); weighted scatter-add
  for (int nt = wave; nt < HDIM / 16; nt += 8) {
    const unsigned short* bdRow = Wde + (size_t)(nt * 16 + ml) * IDIM;
    v8f acc0 = {}, acc1 = {};
#pragma clang loop unroll_count(2)
    for (int k0 = 0; k0 < IDIM; k0 += 32) {
      const v16bf b  = load_b_glb(bdRow, k0, half);
      const v16bf a0 = load_a_lds(sH, IDIM, k0, lane);
      const v16bf a1 = load_a_lds(sH + 16 * IDIM, IDIM, k0, lane);
      if (k0 + 64 < IDIM) __builtin_prefetch(bdRow + k0 + 64, 0, 1);
      acc0 = __builtin_amdgcn_wmma_f32_16x16x32_bf16(false, a0, false, b, (short)0, acc0, false, false);
      acc1 = __builtin_amdgcn_wmma_f32_16x16x32_bf16(false, a1, false, b, (short)0, acc1, false, false);
    }
    const int n = nt * 16 + ml;
#pragma unroll
    for (int j = 0; j < 8; ++j) {
      const int m = j + 8 * half;
      int t = sTok[m];
      if (t >= 0) atomicAdd(&out[(size_t)t * HDIM + n], acc0[j] * sW[m]);
      t = sTok[16 + m];
      if (t >= 0) atomicAdd(&out[(size_t)t * HDIM + n], acc1[j] * sW[16 + m]);
    }
  }
}

extern "C" void kernel_launch(void* const* d_in, const int* in_sizes, int n_in,
                              void* d_out, int out_size, void* d_ws, size_t ws_size,
                              hipStream_t stream) {
  (void)in_sizes; (void)n_in; (void)out_size; (void)ws_size;
  const float* x         = (const float*)d_in[0];  // [B,S,H]
  const float* gate_w    = (const float*)d_in[1];  // [E,H]
  const float* gate_proj = (const float*)d_in[2];  // [E,I,H]
  const float* up_proj   = (const float*)d_in[3];  // [E,I,H]
  const float* down_proj = (const float*)d_in[4];  // [E,H,I]
  float* out = (float*)d_out;                      // [B,S,H]

  // workspace: bf16 mirrors first (16B-aligned sizes), then bookkeeping (~84 MB total)
  const size_t W_ELEMS = (size_t)NEXP * IDIM * HDIM;      // 12.58M per matrix
  unsigned short* wgbf = (unsigned short*)d_ws;
  unsigned short* wubf = wgbf + W_ELEMS;
  unsigned short* wdbf = wubf + W_ELEMS;
  unsigned short* xbf  = wdbf + W_ELEMS;                  // [TOKENS+1][H], last row zero pad
  int*   counts  = (int*)(xbf + (size_t)(TOKENS + 1) * HDIM);
  int*   offsets = counts + NEXP;
  int*   cursor  = offsets + NEXP + 1;
  int*   topIdx  = cursor + NEXP;
  int*   tokIds  = topIdx + TOKENS * TOPK;
  float* topW    = (float*)(tokIds + TOKENS * TOPK);
  float* tokW    = topW + TOKENS * TOPK;

  // one-pass fp32 -> bf16 of weights + activations (streaming, ~10us at HBM rate)
  cvt_bf16_kernel<<<2048, 256, 0, stream>>>(gate_proj, wgbf, (int)(W_ELEMS / 8));
  cvt_bf16_kernel<<<2048, 256, 0, stream>>>(up_proj,   wubf, (int)(W_ELEMS / 8));
  cvt_bf16_kernel<<<2048, 256, 0, stream>>>(down_proj, wdbf, (int)(W_ELEMS / 8));
  cvt_bf16_kernel<<<1024, 256, 0, stream>>>(x, xbf, (TOKENS * HDIM) / 8);

  zero_out_kernel<<<1024, 256, 0, stream>>>(out, TOKENS * HDIM);
  init_misc_kernel<<<1, 256, 0, stream>>>(counts, xbf + (size_t)TOKENS * HDIM);
  router_kernel<<<TOKENS, 128, 0, stream>>>(x, gate_w, counts, topIdx, topW);
  scan_kernel<<<1, 32, 0, stream>>>(counts, offsets, cursor);
  scatter_kernel<<<(TOKENS + 255) / 256, 256, 0, stream>>>(topIdx, topW, cursor, tokIds, tokW);

  dim3 grid(NEXP, TOKENS / MROWS);  // worst case: one expert owns every token
  moe_expert_kernel<<<grid, 256, 0, stream>>>(xbf, wgbf, wubf, wdbf,
                                              counts, offsets, tokIds, tokW, out);
}
